// Attention_42846593744909
// MI455X (gfx1250) — matrix operand order repaired and checked by reading, not compiled, not run on hardware
//
#include <hip/hip_runtime.h>
#include <hip/hip_bf16.h>

// ---------------------------------------------------------------------------
// Types matching the gfx1250 WMMA builtin signatures
// ---------------------------------------------------------------------------
typedef __attribute__((ext_vector_type(16))) __bf16   v16bf;
typedef __attribute__((ext_vector_type(8)))  __bf16   v8bf;
typedef __attribute__((ext_vector_type(8)))  float    v8f;
typedef __attribute__((ext_vector_type(4)))  uint32_t v4u;
typedef __attribute__((ext_vector_type(8)))  uint32_t v8u;

// Problem constants
#define BB 4
#define SS 1024
#define DD 2048
#define HH 32
#define HDIM 64
#define NQKV 96          // H + 2*KVH heads in the fused QKV projection

// ---------------------------------------------------------------------------
// WMMA wrapper:  D(16x16,f32) = A(16x32,bf16) x B(32x16,bf16) + C
// ---------------------------------------------------------------------------
__device__ __forceinline__ v8f wmma_bf16(v16bf a, v16bf b, v8f c) {
    return __builtin_amdgcn_wmma_f32_16x16x32_bf16(
        /*neg_a=*/false, a, /*neg_b=*/false, b,
        /*c_mod=*/(short)0, c, /*reuse_a=*/false, /*reuse_b=*/false);
}

// A-fragment (16x32 bf16, row-major source, K contiguous).
// ISA layout: lane l holds row (l&15); lane-group g = l>>4:
//   halves 0..7  -> K = kk + 8*g + (0..7)
//   halves 8..15 -> K = kk + 16 + 8*g + (0..7)
__device__ __forceinline__ v16bf load_a_frag(const __bf16* rowptr, int kk, int g) {
    v8bf lo = *(const v8bf*)(rowptr + kk + 8 * g);
    v8bf hi = *(const v8bf*)(rowptr + kk + 16 + 8 * g);
    v16bf a;
#pragma unroll
    for (int i = 0; i < 8; ++i) { a[i] = lo[i]; a[i + 8] = hi[i]; }
    return a;
}

// B-fragment (32x16 bf16).  NT GEMM: column n of B == row n of the source
// matrix (K contiguous).  ISA layout: lane l holds col (l&15),
//   halves 0..7  -> K = kk + 8*g + (0..7)
//   halves 8..15 -> K = kk + 16 + 8*g + (0..7)
__device__ __forceinline__ v16bf load_b_frag(const __bf16* colptr, int kk, int g) {
    v8bf lo = *(const v8bf*)(colptr + kk + 8 * g);
    v8bf hi = *(const v8bf*)(colptr + kk + 16 + 8 * g);
    v16bf b;
#pragma unroll
    for (int i = 0; i < 8; ++i) { b[i] = lo[i]; b[i + 8] = hi[i]; }
    return b;
}

// ---------------------------------------------------------------------------
// Tensor Data Mover: 2D tile load Global -> LDS (bf16 elements, data_size=2B).
// D# packed per CDNA5 ISA ch.8 (groups 0/1, VADDR2/3 = NULL -> 2D tensor).
// Descriptor values must be wave-uniform; "s" constraints place them in
// SGPR quad/oct tuples.  Completion tracked with TENSORcnt.
// ---------------------------------------------------------------------------
__device__ __forceinline__ void tdm_load_2d_bf16(uint32_t lds_off, const void* gaddr,
                                                 uint32_t tensor_d0, uint32_t tensor_d1,
                                                 uint32_t tile_d0, uint32_t tile_d1,
                                                 uint64_t stride0_elts) {
    uint64_t ga = (uint64_t)(uintptr_t)gaddr;
    v4u g0;
    g0[0] = 1u;                                            // count=1, user desc
    g0[1] = lds_off;                                       // lds_addr (bytes)
    g0[2] = (uint32_t)ga;                                  // global_addr[31:0]
    g0[3] = (uint32_t)((ga >> 32) & 0x1FFFFFFu)            // global_addr[56:32]
          | (2u << 30);                                    // type = 2 ("image")
    v8u g1;
    g1[0] = (1u << 16);                                    // data_size=1 (2 bytes)
    g1[1] = (tensor_d0 & 0xFFFFu) << 16;                   // tensor_dim0[15:0]
    g1[2] = (tensor_d0 >> 16) | ((tensor_d1 & 0xFFFFu) << 16);
    g1[3] = (tensor_d1 >> 16) | (tile_d0 << 16);           // tile_dim0
    g1[4] = tile_d1 & 0xFFFFu;                             // tile_dim1 (tile_dim2=0)
    g1[5] = (uint32_t)stride0_elts;                        // tensor_dim0_stride lo
    g1[6] = (uint32_t)(stride0_elts >> 32) & 0xFFFFu;      // stride hi (dim1_stride=0)
    g1[7] = 0;
    asm volatile("tensor_load_to_lds %0, %1" :: "s"(g0), "s"(g1) : "memory");
}

// ---------------------------------------------------------------------------
// Kernel 0: f32 -> bf16 conversion (one-time; halves all GEMM read traffic,
// makes every operand L2-resident)
// ---------------------------------------------------------------------------
__global__ void __launch_bounds__(256)
cvt_f32_bf16(const float* __restrict__ in, __bf16* __restrict__ out, int n) {
    int i = blockIdx.x * blockDim.x + threadIdx.x;
    int stride = gridDim.x * blockDim.x;
    for (; i < n; i += stride) out[i] = (__bf16)in[i];
}

// ---------------------------------------------------------------------------
// Kernel 1: fused QKV projection + RoPE.
// Grid: (256 row-tiles of 16 tokens, 24) x 128 threads (4 waves).
// Each wave computes one (16-token x 64-dim) tile = one full qkv head, so the
// RoPE pair (col, col+32) is register-local (C tile t pairs with tile t+2).
// q,k written as (B,H,S,64); v written transposed as (B,H,64,S).
// ---------------------------------------------------------------------------
__global__ void __launch_bounds__(128)
qkv_rope_kernel(const __bf16* __restrict__ hbf,   // (B*S, D) bf16
                const __bf16* __restrict__ wbf,   // (6144, D) bf16
                const float*  __restrict__ cosp,  // (S, 64)
                const float*  __restrict__ sinp,  // (S, 64)
                __bf16* __restrict__ qb,          // (B,H,S,64)
                __bf16* __restrict__ kb,          // (B,H,S,64)
                __bf16* __restrict__ vtb)         // (B,H,64,S)
{
    const int lane = threadIdx.x & 31;
    const int wave = threadIdx.x >> 5;
    const int g  = lane >> 4;
    const int lx = lane & 15;
    const int head    = blockIdx.y * 4 + wave;   // 0..95
    const int rowbase = blockIdx.x * 16;         // token-row tile (B*S rows)

    const __bf16* arow = hbf + (size_t)(rowbase + lx) * DD;
    const __bf16* bcol[4];
#pragma unroll
    for (int t = 0; t < 4; ++t)
        bcol[t] = wbf + (size_t)(head * HDIM + t * 16 + lx) * DD;

    v8f c[4] = {};
    for (int kk = 0; kk < DD; kk += 32) {
        v16bf a = load_a_frag(arow, kk, g);
#pragma unroll
        for (int t = 0; t < 4; ++t)
            c[t] = wmma_bf16(a, load_b_frag(bcol[t], kk, g), c[t]);
    }

    const int b = rowbase >> 10;          // batch
    const int sb = rowbase & (SS - 1);    // seq position of tile row 0

    if (head < 2 * HH) {
        // RoPE on q and k heads: pairs (col, col+32), cos/sin repeat halves.
#pragma unroll
        for (int j = 0; j < 8; ++j) {
            int s = sb + j + 8 * g;
#pragma unroll
            for (int t = 0; t < 2; ++t) {
                int col = t * 16 + lx;                 // 0..31
                float cv = cosp[s * HDIM + col];
                float sv = sinp[s * HDIM + col];
                float lo = c[t][j], hi = c[t + 2][j];
                c[t][j]     = lo * cv - hi * sv;
                c[t + 2][j] = hi * cv + lo * sv;
            }
        }
        __bf16* base = (head < HH ? qb : kb) +
                       (((size_t)b * HH + (head & (HH - 1))) * SS) * HDIM;
#pragma unroll
        for (int t = 0; t < 4; ++t)
#pragma unroll
            for (int j = 0; j < 8; ++j) {
                int s = sb + j + 8 * g;
                base[(size_t)s * HDIM + t * 16 + lx] = (__bf16)c[t][j];
            }
    } else {
        // v head: store transposed (64, S) so PV B-fragments are contiguous
        int h = head - 2 * HH;
        __bf16* base = vtb + (((size_t)b * HH + h) * HDIM) * SS;
#pragma unroll
        for (int t = 0; t < 4; ++t)
#pragma unroll
            for (int j = 0; j < 8; ++j) {
                int s = sb + j + 8 * g;
                base[(size_t)(t * 16 + lx) * SS + s] = (__bf16)c[t][j];
            }
    }
}

// ---------------------------------------------------------------------------
// Kernel 2: flash attention.  Grid (16,128) x 128 threads; 4 waves share one
// (b,h) and process 4 adjacent 16-query tiles.  Per 32-key chunk the K-tile
// (32x64) and V-tile (64x32, from vT) are staged Global->LDS ONCE per block
// by the Tensor Data Mover (wave 0 issues 2 descriptors, s_wait_tensorcnt,
// workgroup barrier), then all 4 waves feed their WMMAs from LDS.
// Online softmax in registers; P converted C-layout -> A-layout via LDS.
// ---------------------------------------------------------------------------
__global__ void __launch_bounds__(128)
flash_attn_kernel(const __bf16* __restrict__ qb,
                  const __bf16* __restrict__ kb,
                  const __bf16* __restrict__ vtb,
                  __bf16* __restrict__ ob)       // (B*S, H*64) bf16
{
    __shared__ __bf16 kt[32 * HDIM];      // [key][dim]   4 KB
    __shared__ __bf16 vt[HDIM * 32];      // [dim][key]   4 KB
    __shared__ __bf16 lds[4 * 16 * 32];   // P staging    4 KB

    const int lane = threadIdx.x & 31;
    const int wave = threadIdx.x >> 5;
    const int g  = lane >> 4;
    const int lx = lane & 15;
    const int bh    = blockIdx.y;                 // 0..127
    const int qtile = blockIdx.x * 4 + wave;      // 0..63
    const int qrow  = qtile * 16;

    const __bf16* qbase = qb  + (size_t)bh * SS * HDIM;
    const __bf16* kbase = kb  + (size_t)bh * SS * HDIM;
    const __bf16* vbase = vtb + (size_t)bh * HDIM * SS;
    __bf16* plds = lds + wave * 16 * 32;

    // LDS byte offsets (generic-pointer low 32 bits == LDS address)
    const uint32_t kt_off = (uint32_t)(uintptr_t)(void*)kt;
    const uint32_t vt_off = (uint32_t)(uintptr_t)(void*)vt;

    const __bf16* qrowp = qbase + (size_t)(qrow + lx) * HDIM;
    v16bf aq0 = load_a_frag(qrowp, 0, g);
    v16bf aq1 = load_a_frag(qrowp, 32, g);

    v8f o[4] = {};
    float m[8], l[8];
#pragma unroll
    for (int j = 0; j < 8; ++j) { m[j] = -INFINITY; l[j] = 0.0f; }

    for (int kv = 0; kv < SS; kv += 32) {
        // fence previous iteration's ds_loads, then refill tiles via TDM
        asm volatile("s_wait_dscnt 0" ::: "memory");
        __syncthreads();
        if (wave == 0) {
            // K tile: rows kv..kv+31 of (S,64) matrix, line length 64
            tdm_load_2d_bf16(kt_off, kbase + (size_t)kv * HDIM,
                             /*tensor_d0=*/HDIM, /*tensor_d1=*/SS - kv,
                             /*tile_d0=*/HDIM, /*tile_d1=*/32,
                             /*stride0=*/HDIM);
            // V tile: 64 dim-rows x 32 keys from vT (64,S), line stride S
            tdm_load_2d_bf16(vt_off, vbase + kv,
                             /*tensor_d0=*/SS - kv, /*tensor_d1=*/HDIM,
                             /*tile_d0=*/32, /*tile_d1=*/HDIM,
                             /*stride0=*/SS);
            __builtin_amdgcn_s_wait_tensorcnt(0);
        }
        __syncthreads();

        // ---- scores: two 16x16 tiles covering 32 keys (K from LDS) -------
        const __bf16* k0 = kt + (size_t)lx * HDIM;
        const __bf16* k1 = kt + (size_t)(16 + lx) * HDIM;
        v8f s0 = {}, s1 = {};
        s0 = wmma_bf16(aq0, load_b_frag(k0, 0, g), s0);
        s0 = wmma_bf16(aq1, load_b_frag(k0, 32, g), s0);
        s1 = wmma_bf16(aq0, load_b_frag(k1, 0, g), s1);
        s1 = wmma_bf16(aq1, load_b_frag(k1, 32, g), s1);

        // ---- online softmax (row stats across a 16-lane group) ----------
        float alpha[8];
#pragma unroll
        for (int j = 0; j < 8; ++j) {
            float a0 = s0[j] * 0.125f;      // 1/sqrt(64)
            float a1 = s1[j] * 0.125f;
            float mx = fmaxf(a0, a1);
            mx = fmaxf(mx, __shfl_xor(mx, 1, 32));
            mx = fmaxf(mx, __shfl_xor(mx, 2, 32));
            mx = fmaxf(mx, __shfl_xor(mx, 4, 32));
            mx = fmaxf(mx, __shfl_xor(mx, 8, 32));
            float mn = fmaxf(m[j], mx);
            alpha[j] = __expf(m[j] - mn);
            m[j] = mn;
            float p0 = __expf(a0 - mn);
            float p1 = __expf(a1 - mn);
            float r = p0 + p1;
            r += __shfl_xor(r, 1, 32);
            r += __shfl_xor(r, 2, 32);
            r += __shfl_xor(r, 4, 32);
            r += __shfl_xor(r, 8, 32);
            l[j] = l[j] * alpha[j] + r;
            // stage P (C-layout: row = j+8g, col = lx / 16+lx) into LDS
            plds[(8 * g + j) * 32 + lx]      = (__bf16)p0;
            plds[(8 * g + j) * 32 + 16 + lx] = (__bf16)p1;
        }
#pragma unroll
        for (int j = 0; j < 8; ++j) {
            o[0][j] *= alpha[j]; o[1][j] *= alpha[j];
            o[2][j] *= alpha[j]; o[3][j] *= alpha[j];
        }
        // per-wave LDS RAW fence before reading P back
        asm volatile("s_wait_dscnt 0" ::: "memory");

        // read P back in A-fragment layout (16x32, ld=32)
        v16bf ap = load_a_frag(plds + lx * 32, 0, g);

        // ---- PV: O(16x64) += P(16x32) x V(32x64)  (V from LDS) ----------
#pragma unroll
        for (int t = 0; t < 4; ++t) {
            const __bf16* vcol = vt + (size_t)(t * 16 + lx) * 32;
            o[t] = wmma_bf16(ap, load_b_frag(vcol, 0, g), o[t]);
        }
    }

    // ---- finalize + store to (B*S, 2048) bf16 ----------------------------
    const int b = bh >> 5, h = bh & (HH - 1);
    __bf16* obase = ob + ((size_t)b * SS + qrow) * (HH * HDIM) + h * HDIM;
#pragma unroll
    for (int j = 0; j < 8; ++j) {
        float inv = 1.0f / l[j];
        int r = j + 8 * g;
#pragma unroll
        for (int t = 0; t < 4; ++t)
            obase[(size_t)r * (HH * HDIM) + t * 16 + lx] = (__bf16)(o[t][j] * inv);
    }
}

// ---------------------------------------------------------------------------
// Kernel 3: output projection.  out(4096x2048,f32) = O(bf16) x w_o^T (NT).
// Grid (256, 8) x 128 threads; one wave -> 16x64 output tile.
// ---------------------------------------------------------------------------
__global__ void __launch_bounds__(128)
out_proj_kernel(const __bf16* __restrict__ obuf,  // (B*S, 2048) bf16
                const __bf16* __restrict__ wob,   // (2048, 2048) bf16
                float* __restrict__ out)          // (B*S, 2048) f32
{
    const int lane = threadIdx.x & 31;
    const int wave = threadIdx.x >> 5;
    const int g  = lane >> 4;
    const int lx = lane & 15;
    const int rowbase = blockIdx.x * 16;
    const int colbase = (blockIdx.y * 4 + wave) * 64;

    const __bf16* arow = obuf + (size_t)(rowbase + lx) * DD;
    const __bf16* bcol[4];
#pragma unroll
    for (int t = 0; t < 4; ++t)
        bcol[t] = wob + (size_t)(colbase + t * 16 + lx) * DD;

    v8f c[4] = {};
    for (int kk = 0; kk < DD; kk += 32) {
        v16bf a = load_a_frag(arow, kk, g);
#pragma unroll
        for (int t = 0; t < 4; ++t)
            c[t] = wmma_bf16(a, load_b_frag(bcol[t], kk, g), c[t]);
    }
#pragma unroll
    for (int t = 0; t < 4; ++t)
#pragma unroll
        for (int j = 0; j < 8; ++j)
            out[(size_t)(rowbase + j + 8 * g) * DD + colbase + t * 16 + lx] = c[t][j];
}

// ---------------------------------------------------------------------------
// Workspace layout (bytes):
//   0         : hidden bf16      (8,388,608 elts)  16,777,216
//   16777216  : w_qkv bf16       (12,582,912)      25,165,824
//   41943040  : w_o bf16         (4,194,304)        8,388,608
//   50331648  : q bf16           (8,388,608)       16,777,216
//   67108864  : k bf16           (8,388,608)       16,777,216
//   83886080  : vT bf16          (8,388,608)       16,777,216
//   100663296 : O bf16           (8,388,608)       16,777,216
//   total: 117,440,512
// ---------------------------------------------------------------------------
extern "C" void kernel_launch(void* const* d_in, const int* in_sizes, int n_in,
                              void* d_out, int out_size, void* d_ws, size_t ws_size,
                              hipStream_t stream) {
    const float* cosp = (const float*)d_in[0];
    const float* sinp = (const float*)d_in[1];
    const float* hid  = (const float*)d_in[2];
    const float* wqkv = (const float*)d_in[3];
    const float* wo   = (const float*)d_in[4];
    float* out = (float*)d_out;

    char* ws = (char*)d_ws;
    __bf16* hbf  = (__bf16*)(ws);
    __bf16* wqbf = (__bf16*)(ws + 16777216);
    __bf16* wobf = (__bf16*)(ws + 41943040);
    __bf16* qb   = (__bf16*)(ws + 50331648);
    __bf16* kb   = (__bf16*)(ws + 67108864);
    __bf16* vtb  = (__bf16*)(ws + 83886080);
    __bf16* obuf = (__bf16*)(ws + 100663296);

    cvt_f32_bf16<<<4096, 256, 0, stream>>>(hid,  hbf,  BB * SS * DD);
    cvt_f32_bf16<<<4096, 256, 0, stream>>>(wqkv, wqbf, NQKV * HDIM * DD);
    cvt_f32_bf16<<<4096, 256, 0, stream>>>(wo,   wobf, DD * HH * HDIM);

    qkv_rope_kernel<<<dim3(256, 24), 128, 0, stream>>>(hbf, wqbf, cosp, sinp,
                                                       qb, kb, vtb);
    flash_attn_kernel<<<dim3(16, 128), 128, 0, stream>>>(qb, kb, vtb, obuf);
    out_proj_kernel<<<dim3(256, 8), 128, 0, stream>>>(obuf, wobf, out);
}
